// MultiLayerGraphRegressionModel_40157944217915
// MI455X (gfx1250) — compile-verified, matrix operand order checked by reading, not run on hardware
//
#include <hip/hip_runtime.h>
#include <hip/hip_bf16.h>
#include <cstdint>

// ---------------------------------------------------------------------------
// Problem constants (match reference)
// ---------------------------------------------------------------------------
#define D_IN  64
#define E_IN  16
#define HCH   128     // hidden width H
#define K_MSG 384     // 3*H
#define K_UPD 256     // 2*H
#define NLAYERS 3
#define NGRAPH  256

typedef __attribute__((ext_vector_type(16))) __bf16 v16bf;
typedef __attribute__((ext_vector_type(8)))  __bf16 v8bf;
typedef __attribute__((ext_vector_type(8)))  float  v8f;

union FragU { v16bf v; struct { v8bf lo; v8bf hi; } p; };

// ---------------------------------------------------------------------------
// CDNA5 async copy: 16 bytes global -> LDS per active lane (ASYNCcnt).
// VDST = LDS byte address (low 32 bits of flat shared pointer, per ISA
// aperture rule); VADDR = 64-bit global address.
// ---------------------------------------------------------------------------
__device__ __forceinline__ void async_cp16(__bf16* lds, const __bf16* g) {
  uint32_t l = (uint32_t)(uintptr_t)lds;
  asm volatile("global_load_async_to_lds_b128 %0, %1, off"
               :: "v"(l), "v"(g) : "memory");
}
__device__ __forceinline__ void async_wait0() {
  asm volatile("s_wait_asynccnt 0x0" ::: "memory");
}

// ---------------------------------------------------------------------------
// WMMA fragment load, 16-bit A/B 16x32 layout (CDNA5 ISA 7.12.2):
//   lane<16 : row = lane,    elems 0..7 -> K=kb+0..7,   elems 8..15 -> K=kb+16..23
//   lane>=16: row = lane-16, elems 0..7 -> K=kb+8..15,  elems 8..15 -> K=kb+24..31
// rowk = (row base pointer) + kb ; hi8 = (lane>>4)*8
// Two 16-byte loads per lane (global_load_b128 or ds_load_b128).
// ---------------------------------------------------------------------------
__device__ __forceinline__ v16bf load_frag(const __bf16* rowk, int hi8) {
  FragU f;
  f.p.lo = *(const v8bf*)(rowk + hi8);
  f.p.hi = *(const v8bf*)(rowk + hi8 + 16);
  return f.v;
}

__device__ __forceinline__ v8f wmma_bf16(v16bf a, v16bf b, v8f c) {
  return __builtin_amdgcn_wmma_f32_16x16x32_bf16(false, a, false, b, (short)0, c,
                                                 false, false);
}

// Store 16x16 f32 C tile with bias + relu -> bf16 row-major [*, HCH]
__device__ __forceinline__ void store_tile_relu(v8f acc, const float* bias,
                                                __bf16* out, int m0, int n0,
                                                int lane) {
  const int r   = lane & 15;
  const int hi8 = (lane >> 4) << 3;
  const float bv = bias[n0 + r];
  __bf16* orow = out + (size_t)(m0 + hi8) * HCH + n0 + r;
#pragma unroll
  for (int i = 0; i < 8; ++i) {
    float v = acc[i] + bv;
    v = fmaxf(v, 0.0f);
    orow[(size_t)i * HCH] = (__bf16)v;
  }
}

// ---------------------------------------------------------------------------
// Generic GEMM: out[M,128] = relu(A[M,Ka] @ WT[128,Ka]^T + bias)
// blockDim = 256 (8 waves), grid.x = M/16. A tile (16 x Ka) staged in LDS via
// async copy once per block; each wave computes one 16x16 WMMA tile.
// ---------------------------------------------------------------------------
__global__ void gemm_bf16_kernel(const __bf16* __restrict__ A,
                                 const __bf16* __restrict__ WT,
                                 const float*  __restrict__ bias,
                                 __bf16* __restrict__ out, int Ka) {
  __shared__ __bf16 sA[16 * HCH];          // Ka <= 128
  const int lane = threadIdx.x & 31;
  const int wave = threadIdx.x >> 5;
  const int m0 = blockIdx.x << 4;
  const int n0 = wave << 4;
  const int r = lane & 15;
  const int hi8 = (lane >> 4) << 3;

  // stage A tile: 2*Ka chunks of 16B
  const int cpr = Ka >> 3;                 // 16B chunks per row
  for (int c = threadIdx.x; c < (Ka << 1); c += 256) {
    int row = c / cpr;
    int col = (c - row * cpr) << 3;
    async_cp16(sA + row * Ka + col, A + (size_t)(m0 + row) * Ka + col);
  }
  async_wait0();
  __syncthreads();

  const __bf16* arow = sA + r * Ka;
  const __bf16* brow = WT + (size_t)(n0 + r) * Ka;
  v8f acc = {};
  for (int kb = 0; kb < Ka; kb += 32)
    acc = wmma_bf16(load_frag(arow + kb, hi8), load_frag(brow + kb, hi8), acc);
  store_tile_relu(acc, bias, out, m0, n0, lane);
}

// ---------------------------------------------------------------------------
// Message GEMM: m[E,128] = relu([h[dst] | h[src] | e] @ WMT[128,384]^T + bM)
// Gathered A rows staged once per block into LDS (16 x 384 bf16 = 12KB).
// ---------------------------------------------------------------------------
__global__ void msg_gemm_kernel(const __bf16* __restrict__ h,
                                const __bf16* __restrict__ e,
                                const int* __restrict__ src,
                                const int* __restrict__ dst,
                                const __bf16* __restrict__ WMT,
                                const float* __restrict__ bias,
                                __bf16* __restrict__ m_out) {
  __shared__ __bf16 sA[16 * K_MSG];
  const int lane = threadIdx.x & 31;
  const int wave = threadIdx.x >> 5;
  const int e0 = blockIdx.x << 4;
  const int n0 = wave << 4;
  const int r = lane & 15;
  const int hi8 = (lane >> 4) << 3;

  // stage: 16 rows x 48 chunks (16B each) = 768 chunks, 3 per thread
#pragma unroll
  for (int it = 0; it < 3; ++it) {
    int c = it * 256 + threadIdx.x;
    int row = c / 48;
    int k = (c - row * 48) << 3;           // element offset within 384
    const __bf16* g;
    if (k < 128)       g = h + (size_t)dst[e0 + row] * HCH + k;
    else if (k < 256)  g = h + (size_t)src[e0 + row] * HCH + (k - 128);
    else               g = e + (size_t)(e0 + row) * HCH + (k - 256);
    async_cp16(sA + row * K_MSG + k, g);
  }
  async_wait0();
  __syncthreads();

  const __bf16* arow = sA + r * K_MSG;
  const __bf16* brow = WMT + (size_t)(n0 + r) * K_MSG;
  v8f acc = {};
#pragma unroll
  for (int kb = 0; kb < K_MSG; kb += 32)
    acc = wmma_bf16(load_frag(arow + kb, hi8), load_frag(brow + kb, hi8), acc);
  store_tile_relu(acc, bias, m_out, e0, n0, lane);
}

// ---------------------------------------------------------------------------
// Update GEMM: h'[N,128] = relu([h | agg] @ WUT[128,256]^T + bU)
// ---------------------------------------------------------------------------
__global__ void update_gemm_kernel(const __bf16* __restrict__ h,
                                   const __bf16* __restrict__ aggb,
                                   const __bf16* __restrict__ WUT,
                                   const float* __restrict__ bias,
                                   __bf16* __restrict__ out) {
  __shared__ __bf16 sA[16 * K_UPD];
  const int lane = threadIdx.x & 31;
  const int wave = threadIdx.x >> 5;
  const int m0 = blockIdx.x << 4;
  const int n0 = wave << 4;
  const int r = lane & 15;
  const int hi8 = (lane >> 4) << 3;

  // stage: 16 rows x 32 chunks = 512 chunks, 2 per thread
#pragma unroll
  for (int it = 0; it < 2; ++it) {
    int c = it * 256 + threadIdx.x;
    int row = c >> 5;
    int k = (c & 31) << 3;                 // element offset within 256
    const __bf16* g = (k < 128) ? (h    + (size_t)(m0 + row) * HCH + k)
                                : (aggb + (size_t)(m0 + row) * HCH + (k - 128));
    async_cp16(sA + row * K_UPD + k, g);
  }
  async_wait0();
  __syncthreads();

  const __bf16* arow = sA + r * K_UPD;
  const __bf16* brow = WUT + (size_t)(n0 + r) * K_UPD;
  v8f acc = {};
#pragma unroll
  for (int kb = 0; kb < K_UPD; kb += 32)
    acc = wmma_bf16(load_frag(arow + kb, hi8), load_frag(brow + kb, hi8), acc);
  store_tile_relu(acc, bias, out, m0, n0, lane);
}

// ---------------------------------------------------------------------------
// Weight transpose+convert: W[K,128] f32 -> WT[128,Kpad] bf16 (zero-pad K)
// ---------------------------------------------------------------------------
__global__ void wconv_kernel(const float* __restrict__ W, __bf16* __restrict__ WT,
                             int K, int Kpad) {
  int i = blockIdx.x * 256 + threadIdx.x;
  int total = Kpad * HCH;
  if (i >= total) return;
  int n = i / Kpad, k = i - n * Kpad;
  WT[(size_t)n * Kpad + k] = (k < K) ? (__bf16)W[(size_t)k * HCH + n] : (__bf16)0.0f;
}

// x [N,64] f32 -> bf16
__global__ void cvt_f32_bf16_kernel(const float* __restrict__ in,
                                    __bf16* __restrict__ out, long total) {
  long i = (long)blockIdx.x * 256 + threadIdx.x;
  if (i < total) out[i] = (__bf16)in[i];
}

// edge_attr [E,16] f32 -> [E,32] bf16 zero-padded
__global__ void pad_edge_kernel(const float* __restrict__ ea,
                                __bf16* __restrict__ eb, int Ecnt) {
  int i = blockIdx.x * 256 + threadIdx.x;
  if (i >= Ecnt * 32) return;
  int e = i >> 5, k = i & 31;
  eb[i] = (k < E_IN) ? (__bf16)ea[(size_t)e * E_IN + k] : (__bf16)0.0f;
}

// segment-sum of messages into agg (f32) + degree counts
__global__ void aggregate_kernel(const __bf16* __restrict__ m,
                                 const int* __restrict__ dst,
                                 float* __restrict__ agg,
                                 float* __restrict__ cnt, int Ecnt) {
  int t = blockIdx.x * 256 + threadIdx.x;
  int edge = t >> 5;
  if (edge >= Ecnt) return;
  int d = dst[edge];
  int c4 = (t & 31) << 2;
  const __bf16* mp = m + (size_t)edge * HCH + c4;
  float* ap = agg + (size_t)d * HCH + c4;
#pragma unroll
  for (int j = 0; j < 4; ++j) atomicAdd(ap + j, (float)mp[j]);
  if ((t & 31) == 0) atomicAdd(cnt + d, 1.0f);
}

// agg f32 / max(cnt,1) -> bf16
__global__ void scale_agg_kernel(const float* __restrict__ agg,
                                 const float* __restrict__ cnt,
                                 __bf16* __restrict__ aggb, long total) {
  long i = (long)blockIdx.x * 256 + threadIdx.x;
  if (i >= total) return;
  float c = fmaxf(cnt[i >> 7], 1.0f);   // H == 128
  aggb[i] = (__bf16)(agg[i] / c);
}

// global mean pool accumulation
__global__ void pool_kernel(const __bf16* __restrict__ h,
                            const int* __restrict__ batch,
                            float* __restrict__ psum,
                            float* __restrict__ pcnt, int Nn) {
  int t = blockIdx.x * 256 + threadIdx.x;
  int node = t >> 5;
  if (node >= Nn) return;
  int g = batch[node];
  int c4 = (t & 31) << 2;
  const __bf16* hp = h + (size_t)node * HCH + c4;
  float* pp = psum + (size_t)g * HCH + c4;
#pragma unroll
  for (int j = 0; j < 4; ++j) atomicAdd(pp + j, (float)hp[j]);
  if ((t & 31) == 0) atomicAdd(pcnt + g, 1.0f);
}

// regression head: out[g] = (psum[g]/cnt) @ Wh + bh
__global__ void head_kernel(const float* __restrict__ psum,
                            const float* __restrict__ pcnt,
                            const float* __restrict__ Wh,
                            const float* __restrict__ bh,
                            float* __restrict__ out) {
  int g = threadIdx.x;   // one block of 256
  float c = fmaxf(pcnt[g], 1.0f);
  float acc = bh[0];
  const float* p = psum + (size_t)g * HCH;
#pragma unroll 4
  for (int k = 0; k < HCH; ++k) acc += (p[k] / c) * Wh[k];
  out[g] = acc;
}

// ---------------------------------------------------------------------------
// Host launcher
// ---------------------------------------------------------------------------
static inline char* ws_take(char*& p, size_t bytes) {
  char* r = p;
  p += (bytes + 255) & ~(size_t)255;
  return r;
}

extern "C" void kernel_launch(void* const* d_in, const int* in_sizes, int n_in,
                              void* d_out, int out_size, void* d_ws, size_t ws_size,
                              hipStream_t stream) {
  const float* x        = (const float*)d_in[0];
  const int*   eidx     = (const int*)d_in[1];
  const float* eattr    = (const float*)d_in[2];
  const int*   batchv   = (const int*)d_in[3];
  const float* Wn0      = (const float*)d_in[4];
  const float* bn0      = (const float*)d_in[5];
  const float* We0      = (const float*)d_in[6];
  const float* be0      = (const float*)d_in[7];
  const float* WM       = (const float*)d_in[8];
  const float* bM       = (const float*)d_in[9];
  const float* WU       = (const float*)d_in[10];
  const float* bU       = (const float*)d_in[11];
  const float* Wn       = (const float*)d_in[12];
  const float* bn       = (const float*)d_in[13];
  const float* We       = (const float*)d_in[14];
  const float* be       = (const float*)d_in[15];
  const float* Wh       = (const float*)d_in[16];
  const float* bh       = (const float*)d_in[17];

  const int Nn = in_sizes[0] / D_IN;      // 100000
  const int Ec = in_sizes[2] / E_IN;      // 640000
  const int* srcI = eidx;                 // edge_index[0]
  const int* dstI = eidx + Ec;            // edge_index[1]

  // ---- workspace carve-up (bf16 activations, f32 accumulators) ----
  char* p = (char*)d_ws;
  __bf16* xb     = (__bf16*)ws_take(p, (size_t)Nn * D_IN * 2);
  __bf16* eapad  = (__bf16*)ws_take(p, (size_t)Ec * 32 * 2);
  __bf16* h_cur  = (__bf16*)ws_take(p, (size_t)Nn * HCH * 2);
  __bf16* h_tmp  = (__bf16*)ws_take(p, (size_t)Nn * HCH * 2);
  __bf16* e_a    = (__bf16*)ws_take(p, (size_t)Ec * HCH * 2);
  __bf16* e_b    = (__bf16*)ws_take(p, (size_t)Ec * HCH * 2);
  __bf16* m_buf  = (__bf16*)ws_take(p, (size_t)Ec * HCH * 2);
  float*  agg    = (float*)ws_take(p, (size_t)Nn * HCH * 4);
  float*  cnt    = (float*)ws_take(p, (size_t)Nn * 4);
  __bf16* aggb   = (__bf16*)ws_take(p, (size_t)Nn * HCH * 2);
  float*  psum   = (float*)ws_take(p, (size_t)NGRAPH * HCH * 4);
  float*  pcnt   = (float*)ws_take(p, (size_t)NGRAPH * 4);
  __bf16* Wn0T   = (__bf16*)ws_take(p, (size_t)HCH * D_IN * 2);
  __bf16* We0T   = (__bf16*)ws_take(p, (size_t)HCH * 32 * 2);
  __bf16* WMT    = (__bf16*)ws_take(p, (size_t)NLAYERS * HCH * K_MSG * 2);
  __bf16* WUT    = (__bf16*)ws_take(p, (size_t)NLAYERS * HCH * K_UPD * 2);
  __bf16* WnT    = (__bf16*)ws_take(p, (size_t)NLAYERS * HCH * HCH * 2);
  __bf16* WeT    = (__bf16*)ws_take(p, (size_t)NLAYERS * HCH * HCH * 2);

  const dim3 blk(256);
  auto cdiv = [](long a, long b) { return (unsigned)((a + b - 1) / b); };

  // ---- weight conversion (transpose to [Nout,K] bf16) ----
  wconv_kernel<<<cdiv(D_IN * HCH, 256), blk, 0, stream>>>(Wn0, Wn0T, D_IN, D_IN);
  wconv_kernel<<<cdiv(32 * HCH, 256), blk, 0, stream>>>(We0, We0T, E_IN, 32);
  for (int l = 0; l < NLAYERS; ++l) {
    wconv_kernel<<<cdiv(K_MSG * HCH, 256), blk, 0, stream>>>(
        WM + (size_t)l * K_MSG * HCH, WMT + (size_t)l * HCH * K_MSG, K_MSG, K_MSG);
    wconv_kernel<<<cdiv(K_UPD * HCH, 256), blk, 0, stream>>>(
        WU + (size_t)l * K_UPD * HCH, WUT + (size_t)l * HCH * K_UPD, K_UPD, K_UPD);
    wconv_kernel<<<cdiv(HCH * HCH, 256), blk, 0, stream>>>(
        Wn + (size_t)l * HCH * HCH, WnT + (size_t)l * HCH * HCH, HCH, HCH);
    wconv_kernel<<<cdiv(HCH * HCH, 256), blk, 0, stream>>>(
        We + (size_t)l * HCH * HCH, WeT + (size_t)l * HCH * HCH, HCH, HCH);
  }

  // ---- input encoders ----
  cvt_f32_bf16_kernel<<<cdiv((long)Nn * D_IN, 256), blk, 0, stream>>>(x, xb, (long)Nn * D_IN);
  pad_edge_kernel<<<cdiv((long)Ec * 32, 256), blk, 0, stream>>>(eattr, eapad, Ec);

  gemm_bf16_kernel<<<Nn / 16, blk, 0, stream>>>(xb, Wn0T, bn0, h_cur, D_IN);
  gemm_bf16_kernel<<<Ec / 16, blk, 0, stream>>>(eapad, We0T, be0, e_a, 32);

  __bf16* e_cur = e_a;
  __bf16* e_nxt = e_b;

  // ---- MPNN layers ----
  for (int l = 0; l < NLAYERS; ++l) {
    msg_gemm_kernel<<<Ec / 16, blk, 0, stream>>>(
        h_cur, e_cur, srcI, dstI, WMT + (size_t)l * HCH * K_MSG, bM + l * HCH, m_buf);

    hipMemsetAsync(agg, 0, (size_t)Nn * HCH * 4, stream);
    hipMemsetAsync(cnt, 0, (size_t)Nn * 4, stream);
    aggregate_kernel<<<cdiv((long)Ec * 32, 256), blk, 0, stream>>>(m_buf, dstI, agg, cnt, Ec);
    scale_agg_kernel<<<cdiv((long)Nn * HCH, 256), blk, 0, stream>>>(agg, cnt, aggb, (long)Nn * HCH);

    update_gemm_kernel<<<Nn / 16, blk, 0, stream>>>(
        h_cur, aggb, WUT + (size_t)l * HCH * K_UPD, bU + l * HCH, h_tmp);
    gemm_bf16_kernel<<<Nn / 16, blk, 0, stream>>>(
        h_tmp, WnT + (size_t)l * HCH * HCH, bn + l * HCH, h_cur, HCH);
    gemm_bf16_kernel<<<Ec / 16, blk, 0, stream>>>(
        e_cur, WeT + (size_t)l * HCH * HCH, be + l * HCH, e_nxt, HCH);
    __bf16* t = e_cur; e_cur = e_nxt; e_nxt = t;
  }

  // ---- global mean pool + head ----
  hipMemsetAsync(psum, 0, (size_t)NGRAPH * HCH * 4, stream);
  hipMemsetAsync(pcnt, 0, (size_t)NGRAPH * 4, stream);
  pool_kernel<<<cdiv((long)Nn * 32, 256), blk, 0, stream>>>(h_cur, batchv, psum, pcnt, Nn);
  head_kernel<<<1, 256, 0, stream>>>(psum, pcnt, Wh, bh, (float*)d_out);

  (void)n_in; (void)out_size; (void)ws_size;
}
